// G_prototype_24378234372465
// MI455X (gfx1250) — matrix operand order repaired and checked by reading, not compiled, not run on hardware
//
#include <hip/hip_runtime.h>
#include <hip/hip_bf16.h>
#include <math.h>

typedef __bf16 bf16;
typedef float        v8f  __attribute__((ext_vector_type(8)));
typedef float        v2f  __attribute__((ext_vector_type(2)));
typedef bf16         v16bf __attribute__((ext_vector_type(16)));
typedef unsigned int v8u  __attribute__((ext_vector_type(8)));
typedef unsigned int v4u  __attribute__((ext_vector_type(4)));

static constexpr int NB  = 8;     // batch
static constexpr int CC  = 256;   // channels
static constexpr int KP  = 32;    // prototypes
static constexpr int WH  = 1024;  // 32*32 pixels
static constexpr int RC3 = 768;   // 3*C
static constexpr int KC  = 8192;  // K*C
static constexpr int KWH = 32768; // K*WH

// A fragment (16x32 bf16, MxK): p -> row + kb + aoff; 16 contiguous k (aoff = 16h); two aligned 16B loads
__device__ __forceinline__ v16bf afrag(const bf16* p) {
    v4u lo = *(const v4u*)p;
    v4u hi = *(const v4u*)(p + 8);
    v8u u = {lo[0], lo[1], lo[2], lo[3], hi[0], hi[1], hi[2], hi[3]};
    return __builtin_bit_cast(v16bf, u);
}
// B fragment (32x16 bf16, KxN): p -> 16 contiguous bf16, 32B aligned
__device__ __forceinline__ v16bf bfrag(const bf16* p) {
    v8u u = *(const v8u*)p;
    return __builtin_bit_cast(v16bf, u);
}
// B fragment with value-select masking (no exec branching)
__device__ __forceinline__ v16bf bfrag_masked(const bf16* p, bool valid) {
    v8u u = *(const v8u*)p;
    v8u z = {0, 0, 0, 0, 0, 0, 0, 0};
    v8u r = valid ? u : z;
    return __builtin_bit_cast(v16bf, r);
}

// ---------------- small utility kernels ----------------

__global__ void k_zero(float* gssc, float* gsss) {
    int t = threadIdx.x;
    if (t < NB) { gssc[t] = 0.f; gsss[t] = 0.f; }
}

// pack conv weights tap-major bf16, t1/t2 weights bf16, centroid channel sums
__global__ void k_pack(const float* __restrict__ conv_w, const float* __restrict__ t1w,
                       const float* __restrict__ t2w, const float* __restrict__ cen,
                       bf16* __restrict__ wpack, bf16* __restrict__ t1wbf,
                       bf16* __restrict__ t2wbf, float* __restrict__ csum) {
    int i = blockIdx.x * blockDim.x + threadIdx.x;
    if (i < 9 * KP * CC) {
        int tap = i / (KP * CC);
        int rem = i % (KP * CC);
        int k = rem / CC, c = rem % CC;
        wpack[i] = (bf16)conv_w[(k * CC + c) * 9 + tap];
    }
    if (i < CC * RC3) t1wbf[i] = (bf16)t1w[i];
    if (i < CC * CC)  t2wbf[i] = (bf16)t2w[i];
    if (i < KP) {
        float s = 0.f;
        for (int c = 0; c < CC; c++) s += cen[i * CC + c];
        csum[i] = s;
    }
}

// L2 normalize feature along C; write channel-major bf16 x, pixel-major bf16 x,
// and per-pixel channel sum
__global__ void k_norm(const float* __restrict__ feat, bf16* __restrict__ xbf,
                       bf16* __restrict__ xbft, float* __restrict__ xsum) {
    int idx = blockIdx.x * blockDim.x + threadIdx.x;   // n*WH + p
    int n = idx >> 10, p = idx & 1023;
    const float* f = feat + n * CC * WH + p;
    float ss = 0.f;
    #pragma unroll 4
    for (int c = 0; c < CC; c++) { float v = f[c * WH]; ss += v * v; }
    float inv = 1.0f / fmaxf(sqrtf(ss), 1e-12f);
    bf16* xo = xbf + n * CC * WH + p;
    bf16* xt = xbft + (size_t)idx * CC;
    float s = 0.f;
    #pragma unroll 4
    for (int c = 0; c < CC; c++) {
        float v = f[c * WH] * inv;
        s += v;
        bf16 bv = (bf16)v;
        xo[c * WH] = bv;
        xt[c] = bv;
    }
    xsum[idx] = s;
}

// ---------------- conv 3x3 as 9 shifted bf16 WMMA GEMMs ----------------
// per wave: one n, one row-aligned 16-pixel tile, both 16-row K_out tiles.
// B from pixel-major x (contiguous 32B fragments, clamped addr + value select)
__global__ void k_conv(const bf16* __restrict__ xbft, const bf16* __restrict__ wpack,
                       const float* __restrict__ conv_b, float* __restrict__ sa) {
    int lane = threadIdx.x & 31, wv = threadIdx.x >> 5;
    int g = blockIdx.x * (blockDim.x >> 5) + wv;   // 0..511
    int n = g >> 6, tile = g & 63;
    int pbase = tile * 16;
    int y = pbase >> 5, xc0 = pbase & 31;
    int col = lane & 15;
    int xc = xc0 + col;
    int koff = (lane < 16) ? 0 : 16;
    int arow = lane & 15, aoff = (lane < 16) ? 0 : 16;
    const bf16* xn = xbft + (size_t)n * WH * CC;
    v8f acc0 = {}, acc1 = {};
    for (int dy = 0; dy < 3; dy++) {
        int yy = y + dy - 1;
        bool vy = (yy >= 0) && (yy < 32);
        int yc = min(max(yy, 0), 31);
        for (int dx = 0; dx < 3; dx++) {
            int xx = xc + dx - 1;
            bool vld = vy && (xx >= 0) && (xx < 32);
            int qc = yc * 32 + min(max(xx, 0), 31);   // always in-bounds
            const bf16* wt = wpack + (dy * 3 + dx) * KP * CC;
            const bf16* bq = xn + (size_t)qc * CC + koff;
            for (int cb = 0; cb < CC; cb += 32) {
                v16bf bfr = bfrag_masked(bq + cb, vld);
                v16bf a0 = afrag(wt + arow * CC + cb + aoff);
                v16bf a1 = afrag(wt + (16 + arow) * CC + cb + aoff);
                acc0 = __builtin_amdgcn_wmma_f32_16x16x32_bf16(false, a0, false, bfr, (short)0, acc0, false, false);
                acc1 = __builtin_amdgcn_wmma_f32_16x16x32_bf16(false, a1, false, bfr, (short)0, acc1, false, false);
            }
        }
    }
    int mo = (lane < 16) ? 0 : 8;
    float* o = sa + n * KP * WH + pbase + col;
    #pragma unroll
    for (int r = 0; r < 8; r++) {
        int m0 = mo + r, m1 = 16 + mo + r;
        o[m0 * WH] = acc0[r] + conv_b[m0];
        o[m1 * WH] = acc1[r] + conv_b[m1];
    }
}

// softmax over K=32 per pixel
__global__ void k_softmax(const float* __restrict__ sa, float* __restrict__ a,
                          bf16* __restrict__ abf) {
    int idx = blockIdx.x * blockDim.x + threadIdx.x;   // n*WH + p
    int n = idx >> 10, p = idx & 1023;
    const float* s = sa + n * KP * WH + p;
    float mx = -1e30f;
    #pragma unroll 4
    for (int k = 0; k < KP; k++) mx = fmaxf(mx, s[k * WH]);
    float den = 0.f;
    #pragma unroll 4
    for (int k = 0; k < KP; k++) den += __expf(s[k * WH] - mx);
    float inv = 1.0f / den;
    float* ao = a + n * KP * WH + p;
    bf16* ab = abf + n * KP * WH + p;
    #pragma unroll 4
    for (int k = 0; k < KP; k++) {
        float v = __expf(s[k * WH] - mx) * inv;
        ao[k * WH] = v;
        ab[k * WH] = (bf16)v;
    }
}

// asum[n,k] = sum_p a
__global__ void k_asum(const float* __restrict__ a, float* __restrict__ asum) {
    int nk = blockIdx.x;
    const float* ap = a + nk * WH;
    float s = 0.f;
    for (int p = threadIdx.x; p < WH; p += 256) s += ap[p];
    __shared__ float red[256];
    red[threadIdx.x] = s; __syncthreads();
    for (int o = 128; o > 0; o >>= 1) {
        if (threadIdx.x < o) red[threadIdx.x] += red[threadIdx.x + o];
        __syncthreads();
    }
    if (threadIdx.x == 0) asum[nk] = red[0];
}

// up_c[n,k,c] = sum_h a[n,k,h]*x[n,c,h] - asum[n,k]*cen[k,c]  (bf16 WMMA, red=WH)
__global__ void k_upc_gemm(const bf16* __restrict__ abf, const bf16* __restrict__ xbf,
                           const float* __restrict__ asum, const float* __restrict__ cen,
                           float* __restrict__ upc) {
    int lane = threadIdx.x & 31, wv = threadIdx.x >> 5;
    int g = blockIdx.x * (blockDim.x >> 5) + wv;   // 0..127
    int n = g >> 4, cb = (g & 15) * 16;
    int col = lane & 15;
    int koff = (lane < 16) ? 0 : 16;
    int arow = lane & 15, aoff = (lane < 16) ? 0 : 16;
    const bf16* an = abf + n * KP * WH;
    const bf16* xn = xbf + n * CC * WH + (cb + col) * WH + koff;
    v8f acc0 = {}, acc1 = {};
    for (int h = 0; h < WH; h += 32) {
        v16bf bfr = bfrag(xn + h);
        v16bf a0 = afrag(an + arow * WH + h + aoff);
        v16bf a1 = afrag(an + (16 + arow) * WH + h + aoff);
        acc0 = __builtin_amdgcn_wmma_f32_16x16x32_bf16(false, a0, false, bfr, (short)0, acc0, false, false);
        acc1 = __builtin_amdgcn_wmma_f32_16x16x32_bf16(false, a1, false, bfr, (short)0, acc1, false, false);
    }
    int mo = (lane < 16) ? 0 : 8;
    int c = cb + col;
    float* o = upc + n * KP * CC + c;
    #pragma unroll
    for (int r = 0; r < 8; r++) {
        int m0 = mo + r, m1 = 16 + mo + r;
        o[m0 * CC] = acc0[r] - asum[n * KP + m0] * cen[m0 * CC + c];
        o[m1 * CC] = acc1[r] - asum[n * KP + m1] * cen[m1 * CC + c];
    }
}

// per-(n,k) L2 over C + accumulate per-n global sumsq
__global__ void k_upc_norm(const float* __restrict__ upc, float* __restrict__ upcn,
                           float* __restrict__ gss) {
    int nk = blockIdx.x, n = nk >> 5;
    float v = upc[nk * CC + threadIdx.x];
    __shared__ float red[256];
    red[threadIdx.x] = v * v; __syncthreads();
    for (int o = 128; o > 0; o >>= 1) {
        if (threadIdx.x < o) red[threadIdx.x] += red[threadIdx.x + o];
        __syncthreads();
    }
    float inv = 1.0f / fmaxf(sqrtf(red[0]), 1e-12f);
    float u = v * inv;
    upcn[nk * CC + threadIdx.x] = u;
    __syncthreads();
    red[threadIdx.x] = u * u; __syncthreads();
    for (int o = 128; o > 0; o >>= 1) {
        if (threadIdx.x < o) red[threadIdx.x] += red[threadIdx.x + o];
        __syncthreads();
    }
    if (threadIdx.x == 0) atomicAdd(&gss[n], red[0]);
}

// up_s[n,k,p] = a*(xsum-csum); per-(n,k) L2 over WH; accumulate per-n sumsq
__global__ void k_ups_norm(const float* __restrict__ a, const float* __restrict__ xsum,
                           const float* __restrict__ csum, float* __restrict__ upsn,
                           float* __restrict__ gss) {
    int nk = blockIdx.x, n = nk >> 5, k = nk & 31;
    const float* ap = a + nk * WH;
    const float* xs = xsum + n * WH;
    float cs = csum[k];
    float vals[4], ss = 0.f;
    #pragma unroll
    for (int i = 0; i < 4; i++) {
        int p = threadIdx.x + i * 256;
        float v = ap[p] * (xs[p] - cs);
        vals[i] = v; ss += v * v;
    }
    __shared__ float red[256];
    red[threadIdx.x] = ss; __syncthreads();
    for (int o = 128; o > 0; o >>= 1) {
        if (threadIdx.x < o) red[threadIdx.x] += red[threadIdx.x + o];
        __syncthreads();
    }
    float inv = 1.0f / fmaxf(sqrtf(red[0]), 1e-12f);
    __syncthreads();
    float ss2 = 0.f;
    #pragma unroll
    for (int i = 0; i < 4; i++) {
        float u = vals[i] * inv;
        upsn[nk * WH + threadIdx.x + i * 256] = u;
        ss2 += u * u;
    }
    red[threadIdx.x] = ss2; __syncthreads();
    for (int o = 128; o > 0; o >>= 1) {
        if (threadIdx.x < o) red[threadIdx.x] += red[threadIdx.x + o];
        __syncthreads();
    }
    if (threadIdx.x == 0) atomicAdd(&gss[n], red[0]);
}

// generic fp32 FC via V_WMMA_F32_16X16X4_F32: out[m,o] = scale_m * A[m,:] . W[o,:] + b[o]
// A is (8 x red) fp32 (rows 8..15 zero-padded); W is (outw x red) fp32 row-major
__global__ void k_fc_f32(const float* __restrict__ A, const float* __restrict__ gss,
                         const float* __restrict__ w, const float* __restrict__ b,
                         float* __restrict__ outv, int red, int outw) {
    int lane = threadIdx.x & 31, wv = threadIdx.x >> 5;
    int g = blockIdx.x * (blockDim.x >> 5) + wv;
    int ob = g * 16;
    int col = lane & 15;
    int arow = lane & 15;
    bool am = arow < NB;
    float asc = 1.0f / fmaxf(sqrtf(gss[arow & 7]), 1e-12f);
    int kk = (lane < 16) ? 0 : 2;
    const float* arp = A + arow * red + kk;
    const float* bcp = w + (ob + col) * red;
    const float* bkp = bcp + kk;
    v8f acc = {};
    for (int k0 = 0; k0 < red; k0 += 256) {
        __builtin_prefetch(bcp + k0 + 1024, 0, 1);
        for (int k = k0; k < k0 + 256; k += 4) {
            v2f af, bfv;
            float a0 = am ? arp[k] : 0.f;
            float a1 = am ? arp[k + 1] : 0.f;
            af[0] = a0 * asc;
            af[1] = a1 * asc;
            bfv[0] = bkp[k];
            bfv[1] = bkp[k + 1];
            acc = __builtin_amdgcn_wmma_f32_16x16x4_f32(false, af, false, bfv, (short)0, acc, false, false);
        }
    }
    int mo = (lane < 16) ? 0 : 8;
    float bias = b[ob + col];
    #pragma unroll
    for (int r = 0; r < 8; r++) {
        int m = mo + r;
        if (m < NB) outv[m * outw + ob + col] = acc[r] + bias;
    }
}

// build gated concat new_t[n][p][3C] in bf16 (pixel-major)
__global__ void k_newt(const float* __restrict__ feat, const float* __restrict__ upcv,
                       const float* __restrict__ upsv, bf16* __restrict__ newt) {
    int idx = blockIdx.x * blockDim.x + threadIdx.x;   // n*WH + p
    int n = idx >> 10, p = idx & 1023;
    const float* f = feat + n * CC * WH + p;
    float ups = upsv[idx];
    const float* uc = upcv + n * CC;
    bf16* o = newt + (size_t)idx * RC3;
    #pragma unroll 4
    for (int c = 0; c < CC; c++) {
        float v = f[c * WH];
        o[c] = (bf16)v;
        o[CC + c] = (bf16)(v * uc[c]);
        o[2 * CC + c] = (bf16)(v * ups);
    }
}

// t1: h1_t[n][p][o] = relu( sum_r t1_w[o,r] * new_t[n][p][r] + b[o] )
__global__ void k_t1(const bf16* __restrict__ newt, const bf16* __restrict__ t1wbf,
                     const float* __restrict__ t1b, bf16* __restrict__ h1t) {
    int lane = threadIdx.x & 31, wv = threadIdx.x >> 5;
    int g = blockIdx.x * (blockDim.x >> 5) + wv;   // 0..2047
    int n = g >> 8, rest = g & 255;
    int mbase = (rest >> 6) * 64;
    int pbase = (rest & 63) * 16;
    int col = lane & 15;
    int p = pbase + col;
    int koff = (lane < 16) ? 0 : 16;
    int arow = lane & 15, aoff = (lane < 16) ? 0 : 16;
    const bf16* bn = newt + (size_t)(n * WH + p) * RC3 + koff;
    v8f acc[4] = {{}, {}, {}, {}};
    for (int kb = 0; kb < RC3; kb += 32) {
        v16bf bfr = bfrag(bn + kb);
        #pragma unroll
        for (int t = 0; t < 4; t++) {
            v16bf af = afrag(t1wbf + (mbase + t * 16 + arow) * RC3 + kb + aoff);
            acc[t] = __builtin_amdgcn_wmma_f32_16x16x32_bf16(false, af, false, bfr, (short)0, acc[t], false, false);
        }
    }
    int mo = (lane < 16) ? 0 : 8;
    #pragma unroll
    for (int t = 0; t < 4; t++) {
        bf16* o = h1t + (size_t)(n * WH + p) * CC + mbase + t * 16 + mo;
        #pragma unroll
        for (int r = 0; r < 8; r++) {
            float v = acc[t][r] + t1b[mbase + t * 16 + mo + r];
            o[r] = (bf16)fmaxf(v, 0.0f);
        }
    }
}

// t2: out[n,o,p] = relu( sum_c t2_w[o,c] * h1_t[n][p][c] + b[o] ), fp32 output
__global__ void k_t2(const bf16* __restrict__ h1t, const bf16* __restrict__ t2wbf,
                     const float* __restrict__ t2b, float* __restrict__ out) {
    int lane = threadIdx.x & 31, wv = threadIdx.x >> 5;
    int g = blockIdx.x * (blockDim.x >> 5) + wv;   // 0..2047
    int n = g >> 8, rest = g & 255;
    int mbase = (rest >> 6) * 64;
    int pbase = (rest & 63) * 16;
    int col = lane & 15;
    int p = pbase + col;
    int koff = (lane < 16) ? 0 : 16;
    int arow = lane & 15, aoff = (lane < 16) ? 0 : 16;
    const bf16* bn = h1t + (size_t)(n * WH + p) * CC + koff;
    v8f acc[4] = {{}, {}, {}, {}};
    for (int kb = 0; kb < CC; kb += 32) {
        v16bf bfr = bfrag(bn + kb);
        #pragma unroll
        for (int t = 0; t < 4; t++) {
            v16bf af = afrag(t2wbf + (mbase + t * 16 + arow) * CC + kb + aoff);
            acc[t] = __builtin_amdgcn_wmma_f32_16x16x32_bf16(false, af, false, bfr, (short)0, acc[t], false, false);
        }
    }
    int mo = (lane < 16) ? 0 : 8;
    #pragma unroll
    for (int t = 0; t < 4; t++) {
        float* o = out + (size_t)(n * CC + mbase + t * 16) * WH + p;
        #pragma unroll
        for (int r = 0; r < 8; r++) {
            float v = acc[t][r] + t2b[mbase + t * 16 + mo + r];
            o[(mo + r) * WH] = fmaxf(v, 0.0f);
        }
    }
}

extern "C" void kernel_launch(void* const* d_in, const int* in_sizes, int n_in,
                              void* d_out, int out_size, void* d_ws, size_t ws_size,
                              hipStream_t stream) {
    const float* feature = (const float*)d_in[0];
    const float* conv_w  = (const float*)d_in[1];
    const float* conv_b  = (const float*)d_in[2];
    const float* cen     = (const float*)d_in[3];
    const float* upfc_w  = (const float*)d_in[4];
    const float* upfc_b  = (const float*)d_in[5];
    const float* upfs_w  = (const float*)d_in[6];
    const float* upfs_b  = (const float*)d_in[7];
    const float* t1_w    = (const float*)d_in[8];
    const float* t1_b    = (const float*)d_in[9];
    const float* t2_w    = (const float*)d_in[10];
    const float* t2_b    = (const float*)d_in[11];
    float* out = (float*)d_out;

    char* wsp = (char*)d_ws;
    auto alloc = [&](size_t bytes) -> char* {
        char* p = wsp;
        wsp += (bytes + 255) & ~(size_t)255;
        return p;
    };
    bf16*  xbf   = (bf16*)alloc((size_t)NB * CC * WH * 2);   // channel-major x
    bf16*  xbft  = (bf16*)alloc((size_t)NB * WH * CC * 2);   // pixel-major x
    float* xsum  = (float*)alloc((size_t)NB * WH * 4);
    float* sa    = (float*)alloc((size_t)NB * KP * WH * 4);
    float* a     = (float*)alloc((size_t)NB * KP * WH * 4);
    bf16*  abf   = (bf16*)alloc((size_t)NB * KP * WH * 2);
    float* asum  = (float*)alloc((size_t)NB * KP * 4);
    float* csum  = (float*)alloc((size_t)KP * 4);
    bf16*  wpack = (bf16*)alloc((size_t)9 * KP * CC * 2);
    bf16*  t1wbf = (bf16*)alloc((size_t)CC * RC3 * 2);
    bf16*  t2wbf = (bf16*)alloc((size_t)CC * CC * 2);
    float* upc   = (float*)alloc((size_t)NB * KP * CC * 4);
    float* upcn  = (float*)alloc((size_t)NB * KP * CC * 4);
    float* gssc  = (float*)alloc((size_t)NB * 4);
    float* gsss  = (float*)alloc((size_t)NB * 4);
    float* upcv  = (float*)alloc((size_t)NB * CC * 4);
    float* upsn  = (float*)alloc((size_t)NB * KP * WH * 4);
    float* upsv  = (float*)alloc((size_t)NB * WH * 4);
    bf16*  newt  = (bf16*)alloc((size_t)NB * WH * RC3 * 2);  // pixel-major concat
    bf16*  h1t   = (bf16*)alloc((size_t)NB * WH * CC * 2);   // pixel-major h1

    k_zero<<<1, 64, 0, stream>>>(gssc, gsss);
    k_pack<<<768, 256, 0, stream>>>(conv_w, t1_w, t2_w, cen, wpack, t1wbf, t2wbf, csum);
    k_norm<<<32, 256, 0, stream>>>(feature, xbf, xbft, xsum);
    k_conv<<<64, 256, 0, stream>>>(xbft, wpack, conv_b, sa);
    k_softmax<<<32, 256, 0, stream>>>(sa, a, abf);
    k_asum<<<256, 256, 0, stream>>>(a, asum);
    k_upc_gemm<<<16, 256, 0, stream>>>(abf, xbf, asum, cen, upc);
    k_upc_norm<<<256, 256, 0, stream>>>(upc, upcn, gssc);
    k_ups_norm<<<256, 256, 0, stream>>>(a, xsum, csum, upsn, gsss);
    k_fc_f32<<<2, 256, 0, stream>>>(upcn, gssc, upfc_w, upfc_b, upcv, KC, CC);
    k_fc_f32<<<8, 256, 0, stream>>>(upsn, gsss, upfs_w, upfs_b, upsv, KWH, WH);
    k_newt<<<32, 256, 0, stream>>>(feature, upcv, upsv, newt);
    k_t1<<<256, 256, 0, stream>>>(newt, t1wbf, t1_b, h1t);
    k_t2<<<256, 256, 0, stream>>>(h1t, t2wbf, t2_b, out);
}
